// Decoder_25220047962694
// MI455X (gfx1250) — compile-verified, matrix operand order checked by reading, not run on hardware
//
#include <hip/hip_runtime.h>

#define ATOM_DIM 34
#define HID      256
#define N_LAYERS 2
#define N_HEADS  8
#define PF_DIM   1024
#define BATCH    32
#define NA       512
#define NP       1024

typedef __attribute__((ext_vector_type(16))) __bf16 v16bf;
typedef __attribute__((ext_vector_type(8)))  __bf16 v8bf;
typedef __attribute__((ext_vector_type(8)))  float  v8f;

__device__ __forceinline__ v8f wmma_bf16(v16bf a, v16bf b, v8f c) {
    // D = A(16x32 bf16) x B(32x16 bf16) + C(16x16 f32)
    return __builtin_amdgcn_wmma_f32_16x16x32_bf16(
        false, a, false, b, (short)0, c, false, false);
}

// Fragment layout (16-bit A, and symmetric B): element e<8 -> k = half*8+e,
// e>=8 -> k = 16+half*8+(e-8).  p = row base + kbase (bf16 elements).
// Two contiguous 16B loads, no conversion.
__device__ __forceinline__ v16bf frag_ld_bf16(const __bf16* __restrict__ p, int half) {
    v8bf lo = *(const v8bf*)(p + half * 8);
    v8bf hi = *(const v8bf*)(p + 16 + half * 8);
    v16bf r;
#pragma unroll
    for (int j = 0; j < 8; ++j) { r[j] = lo[j]; r[8 + j] = hi[j]; }
    return r;
}

// fp32 source variant (used for the softmax-probability tile held in LDS).
__device__ __forceinline__ v16bf frag_ld_f32(const float* __restrict__ p, int half) {
    v16bf r;
    const float* p0 = p + half * 8;
#pragma unroll
    for (int j = 0; j < 8; ++j) {
        r[j]     = (__bf16)p0[j];
        r[8 + j] = (__bf16)p0[16 + j];
    }
    return r;
}

// Strided bf16 gather: value(e) = p[k(e) * stride] (V columns in attention).
__device__ __forceinline__ v16bf frag_ld_bf16_strided(const __bf16* __restrict__ p,
                                                      int stride, int half) {
    v16bf r;
    const __bf16* p0 = p + (size_t)(half * 8) * stride;
#pragma unroll
    for (int j = 0; j < 8; ++j) {
        r[j]     = p0[(size_t)j * stride];
        r[8 + j] = p0[(size_t)(16 + j) * stride];
    }
    return r;
}

// ---------------------------------------------------------------------------
// Bulk fp32 -> bf16 conversion (weights / src, once per launch)
// ---------------------------------------------------------------------------
__global__ __launch_bounds__(256) void cvt_bf16(const float* __restrict__ in,
                                                __bf16* __restrict__ out, int n) {
    int i = blockIdx.x * 256 + threadIdx.x;
    if (i < n) out[i] = (__bf16)in[i];
}

// ---------------------------------------------------------------------------
// C[M,N] = act(A[M,K] @ W[N,K]^T + bias[N])
// A, W bf16; accumulate fp32; output fp32 or bf16 (template).
// Block: 256 threads = 8 waves; block tile 64x128; wave tile 32x32.
// ---------------------------------------------------------------------------
template <bool RELU, bool OUTBF>
__global__ __launch_bounds__(256) void gemm_bias(
    const __bf16* __restrict__ A, const __bf16* __restrict__ W,
    const float* __restrict__ bias, void* __restrict__ C,
    int M, int N, int K) {
    const int lane = threadIdx.x & 31;
    const int wid  = threadIdx.x >> 5;
    const int half = lane >> 4;
    const int lrow = lane & 15;
    const int mbase = blockIdx.y * 64  + (wid & 1) * 32;
    const int nbase = blockIdx.x * 128 + (wid >> 1) * 32;

    v8f acc[2][2] = {};
    for (int kk = 0; kk < K; kk += 32) {
        v16bf af[2], bf[2];
#pragma unroll
        for (int mt = 0; mt < 2; ++mt)
            af[mt] = frag_ld_bf16(A + (size_t)(mbase + mt * 16 + lrow) * K + kk, half);
#pragma unroll
        for (int nt = 0; nt < 2; ++nt)
            bf[nt] = frag_ld_bf16(W + (size_t)(nbase + nt * 16 + lrow) * K + kk, half);
#pragma unroll
        for (int mt = 0; mt < 2; ++mt)
#pragma unroll
            for (int nt = 0; nt < 2; ++nt)
                acc[mt][nt] = wmma_bf16(af[mt], bf[nt], acc[mt][nt]);
    }

#pragma unroll
    for (int mt = 0; mt < 2; ++mt)
#pragma unroll
        for (int nt = 0; nt < 2; ++nt) {
            const int col = nbase + nt * 16 + lrow;
            const float bv = bias[col];
#pragma unroll
            for (int r = 0; r < 8; ++r) {
                const int row = mbase + mt * 16 + r + half * 8;
                float v = acc[mt][nt][r] + bv;
                if (RELU) v = fmaxf(v, 0.0f);
                if (OUTBF) ((__bf16*)C)[(size_t)row * N + col] = (__bf16)v;
                else       ((float*)C)[(size_t)row * N + col]  = v;
            }
        }
}

// ---------------------------------------------------------------------------
// Attention: one (batch, head, 16-query tile) per wave.  d = 32 == WMMA K.
// q/out: [B, NA, 256] bf16, head h at cols h*32..; k/v: [B, LK, 256] bf16.
// ---------------------------------------------------------------------------
template <int LK>
__global__ __launch_bounds__(32) void attention(
    const __bf16* __restrict__ q, const __bf16* __restrict__ k,
    const __bf16* __restrict__ v, __bf16* __restrict__ out) {
    __shared__ float s[16 * LK];

    const int lane = threadIdx.x & 31;
    const int half = lane >> 4;
    const int lrow = lane & 15;
    const int qt = blockIdx.x;
    const int h  = blockIdx.y;
    const int b  = blockIdx.z;

    const size_t qoff  = ((size_t)b * NA + (size_t)qt * 16) * HID + h * 32;
    const size_t kvoff = ((size_t)b * LK) * HID + h * 32;
    const float scale  = 0.17677669529663687f;  // 1/sqrt(32)

    const v16bf qf = frag_ld_bf16(q + qoff + (size_t)lrow * HID, half);

    // Pass 1: S = (Q K^T) * scale -> LDS
    for (int kt = 0; kt < LK; kt += 16) {
        v16bf kf = frag_ld_bf16(k + kvoff + (size_t)(kt + lrow) * HID, half);
        v8f sc = {};
        sc = wmma_bf16(qf, kf, sc);
#pragma unroll
        for (int r = 0; r < 8; ++r)
            s[(r + half * 8) * LK + kt + lrow] = sc[r] * scale;
    }
    __syncthreads();

    // Softmax per query row (lanes 0..15 own one row each)
    if (lane < 16) {
        float* row = s + lane * LK;
        float mx = -3.402823466e38f;
        for (int j = 0; j < LK; ++j) mx = fmaxf(mx, row[j]);
        float sum = 0.0f;
        for (int j = 0; j < LK; ++j) { float e = __expf(row[j] - mx); row[j] = e; sum += e; }
        const float inv = 1.0f / sum;
        for (int j = 0; j < LK; ++j) row[j] *= inv;
    }
    __syncthreads();

    // Pass 2: O = P @ V  (two 16-col accumulators cover d = 32)
    v8f o0 = {}, o1 = {};
    for (int kt = 0; kt < LK; kt += 32) {
        v16bf pf = frag_ld_f32(&s[lrow * LK + kt], half);
        v16bf v0 = frag_ld_bf16_strided(v + kvoff + (size_t)kt * HID + lrow,      HID, half);
        v16bf v1 = frag_ld_bf16_strided(v + kvoff + (size_t)kt * HID + 16 + lrow, HID, half);
        o0 = wmma_bf16(pf, v0, o0);
        o1 = wmma_bf16(pf, v1, o1);
    }
#pragma unroll
    for (int r = 0; r < 8; ++r) {
        const size_t row = (size_t)b * NA + (size_t)qt * 16 + r + half * 8;
        out[row * HID + h * 32 + lrow]      = (__bf16)o0[r];
        out[row * HID + h * 32 + 16 + lrow] = (__bf16)o1[r];
    }
}

// ---------------------------------------------------------------------------
// x = LN(x + r) * g + b; also writes bf16 mirror xb for the next GEMMs.
// ---------------------------------------------------------------------------
__global__ __launch_bounds__(256) void add_ln(
    float* __restrict__ x, const float* __restrict__ r,
    const float* __restrict__ g, const float* __restrict__ beta,
    __bf16* __restrict__ xb) {
    __shared__ float red[256];
    const int row = blockIdx.x, i = threadIdx.x;
    const size_t idx = (size_t)row * HID + i;
    const float val = x[idx] + r[idx];

    red[i] = val;
    __syncthreads();
    for (int st = 128; st > 0; st >>= 1) {
        if (i < st) red[i] += red[i + st];
        __syncthreads();
    }
    const float mean = red[0] * (1.0f / HID);
    __syncthreads();

    const float d = val - mean;
    red[i] = d * d;
    __syncthreads();
    for (int st = 128; st > 0; st >>= 1) {
        if (i < st) red[i] += red[i + st];
        __syncthreads();
    }
    const float var = red[0] * (1.0f / HID);
    __syncthreads();

    const float y = d * rsqrtf(var + 1e-5f) * g[i] + beta[i];
    x[idx]  = y;
    xb[idx] = (__bf16)y;
}

// ---------------------------------------------------------------------------
// x0 = trg @ ft_w^T + ft_b  (K = 34, scalar); fp32 + bf16 mirror.
// ---------------------------------------------------------------------------
__global__ __launch_bounds__(256) void featurize(
    const float* __restrict__ trg, const float* __restrict__ w,
    const float* __restrict__ bias, float* __restrict__ outf,
    __bf16* __restrict__ outb) {
    __shared__ float t[ATOM_DIM];
    const int row = blockIdx.x, n = threadIdx.x;
    if (n < ATOM_DIM) t[n] = trg[(size_t)row * ATOM_DIM + n];
    __syncthreads();
    float acc = bias[n];
#pragma unroll
    for (int j = 0; j < ATOM_DIM; ++j) acc += t[j] * w[n * ATOM_DIM + j];
    outf[(size_t)row * HID + n] = acc;
    outb[(size_t)row * HID + n] = (__bf16)acc;
}

// ---------------------------------------------------------------------------
// Norm-weighted softmax pooling over NA rows (one batch per block)
// ---------------------------------------------------------------------------
__global__ __launch_bounds__(256) void pool(
    const float* __restrict__ x, float* __restrict__ pooled) {
    __shared__ float wrow[NA];
    __shared__ float red[256];
    const int b = blockIdx.x, t = threadIdx.x;
    const float* xb = x + (size_t)b * NA * HID;

    for (int r = t; r < NA; r += 256) {
        float s = 0.0f;
        for (int j = 0; j < HID; ++j) { float v = xb[(size_t)r * HID + j]; s += v * v; }
        wrow[r] = sqrtf(s);
    }
    __syncthreads();

    red[t] = fmaxf(wrow[t], wrow[t + 256]);
    __syncthreads();
    for (int st = 128; st > 0; st >>= 1) {
        if (t < st) red[t] = fmaxf(red[t], red[t + st]);
        __syncthreads();
    }
    const float mx = red[0];
    __syncthreads();

    const float e0 = __expf(wrow[t] - mx);
    const float e1 = __expf(wrow[t + 256] - mx);
    wrow[t] = e0; wrow[t + 256] = e1;
    red[t] = e0 + e1;
    __syncthreads();
    for (int st = 128; st > 0; st >>= 1) {
        if (t < st) red[t] += red[t + st];
        __syncthreads();
    }
    const float inv = 1.0f / red[0];
    __syncthreads();

    float acc = 0.0f;
    for (int r = 0; r < NA; ++r) acc += wrow[r] * xb[(size_t)r * HID + t];
    pooled[(size_t)b * HID + t] = acc * inv;
}

// ---------------------------------------------------------------------------
// out = relu(pooled @ fc1^T + b1) @ fc2^T + b2   (tiny, one batch per block)
// ---------------------------------------------------------------------------
__global__ __launch_bounds__(256) void head(
    const float* __restrict__ pooled,
    const float* __restrict__ w1, const float* __restrict__ b1,
    const float* __restrict__ w2, const float* __restrict__ b2,
    float* __restrict__ out) {
    __shared__ float p[HID];
    __shared__ float h1[HID];
    const int b = blockIdx.x, t = threadIdx.x;
    p[t] = pooled[(size_t)b * HID + t];
    __syncthreads();
    float acc = b1[t];
    for (int j = 0; j < HID; ++j) acc += p[j] * w1[t * HID + j];
    h1[t] = fmaxf(acc, 0.0f);
    __syncthreads();
    if (t < 2) {
        float a2 = b2[t];
        for (int j = 0; j < HID; ++j) a2 += h1[j] * w2[t * HID + j];
        out[(size_t)b * 2 + t] = a2;
    }
}

// ---------------------------------------------------------------------------
extern "C" void kernel_launch(void* const* d_in, const int* in_sizes, int n_in,
                              void* d_out, int out_size, void* d_ws, size_t ws_size,
                              hipStream_t stream) {
    (void)in_sizes; (void)n_in; (void)out_size; (void)ws_size;
    const float* trg   = (const float*)d_in[0];
    const float* src   = (const float*)d_in[1];
    const float* ft_w  = (const float*)d_in[2];
    const float* ft_b  = (const float*)d_in[3];
    const float* ln_g  = (const float*)d_in[4];
    const float* ln_b  = (const float*)d_in[5];
    const float* sa_wq = (const float*)d_in[6];  const float* sa_bq = (const float*)d_in[7];
    const float* sa_wk = (const float*)d_in[8];  const float* sa_bk = (const float*)d_in[9];
    const float* sa_wv = (const float*)d_in[10]; const float* sa_bv = (const float*)d_in[11];
    const float* sa_wf = (const float*)d_in[12]; const float* sa_bf = (const float*)d_in[13];
    const float* ea_wq = (const float*)d_in[14]; const float* ea_bq = (const float*)d_in[15];
    const float* ea_wk = (const float*)d_in[16]; const float* ea_bk = (const float*)d_in[17];
    const float* ea_wv = (const float*)d_in[18]; const float* ea_bv = (const float*)d_in[19];
    const float* ea_wf = (const float*)d_in[20]; const float* ea_bf = (const float*)d_in[21];
    const float* pf_w1 = (const float*)d_in[22]; const float* pf_b1 = (const float*)d_in[23];
    const float* pf_w2 = (const float*)d_in[24]; const float* pf_b2 = (const float*)d_in[25];
    const float* fc1_w = (const float*)d_in[26]; const float* fc1_b = (const float*)d_in[27];
    const float* fc2_w = (const float*)d_in[28]; const float* fc2_b = (const float*)d_in[29];
    float* out = (float*)d_out;

    const size_t M  = (size_t)BATCH * NA;   // 16384 token rows
    const size_t MS = (size_t)BATCH * NP;   // 32768 source rows

    // ---- workspace layout ----
    float*  x   = (float*)d_ws;             // fp32 master x     [M,256]
    float*  tm  = x  + M * HID;             // fp32 pre-LN tmp   [M,256]
    float*  pld = tm + M * HID;             // pooled            [32,256]
    __bf16* xb  = (__bf16*)(pld + BATCH * HID);  // bf16 mirror of x
    __bf16* qb  = xb  + M * HID;
    __bf16* kb  = qb  + M * HID;            // [MS,256] (self uses first M rows)
    __bf16* vb  = kb  + MS * HID;
    __bf16* ab  = vb  + MS * HID;           // attention output  [M,256]
    __bf16* hb  = ab  + M * HID;            // FFN hidden        [M,1024]
    __bf16* srcb = hb + M * PF_DIM;         // bf16 src          [MS,256]
    __bf16* wb  = srcb + MS * HID;          // bf16 weights

    // bf16 weight pool offsets (elements); each proj tensor is [L,256,256]
    const size_t PW = (size_t)HID * HID;           // 65536 per layer
    __bf16* wq_b = wb;                             // sa_wq  2*PW
    __bf16* wk_b = wq_b + N_LAYERS * PW;
    __bf16* wv_b = wk_b + N_LAYERS * PW;
    __bf16* wf_b = wv_b + N_LAYERS * PW;
    __bf16* eq_b = wf_b + N_LAYERS * PW;
    __bf16* ek_b = eq_b + N_LAYERS * PW;
    __bf16* ev_b = ek_b + N_LAYERS * PW;
    __bf16* ef_b = ev_b + N_LAYERS * PW;
    __bf16* w1_b = ef_b + N_LAYERS * PW;           // [L,1024,256]
    __bf16* w2_b = w1_b + (size_t)N_LAYERS * PF_DIM * HID;  // [L,256,1024]

    // ---- one-time conversions (weights + src) ----
    const int nProj = N_LAYERS * (int)PW;              // 131072
    const int nFfn  = N_LAYERS * PF_DIM * HID;         // 524288
    cvt_bf16<<<nProj / 256, 256, 0, stream>>>(sa_wq, wq_b, nProj);
    cvt_bf16<<<nProj / 256, 256, 0, stream>>>(sa_wk, wk_b, nProj);
    cvt_bf16<<<nProj / 256, 256, 0, stream>>>(sa_wv, wv_b, nProj);
    cvt_bf16<<<nProj / 256, 256, 0, stream>>>(sa_wf, wf_b, nProj);
    cvt_bf16<<<nProj / 256, 256, 0, stream>>>(ea_wq, eq_b, nProj);
    cvt_bf16<<<nProj / 256, 256, 0, stream>>>(ea_wk, ek_b, nProj);
    cvt_bf16<<<nProj / 256, 256, 0, stream>>>(ea_wv, ev_b, nProj);
    cvt_bf16<<<nProj / 256, 256, 0, stream>>>(ea_wf, ef_b, nProj);
    cvt_bf16<<<nFfn  / 256, 256, 0, stream>>>(pf_w1, w1_b, nFfn);
    cvt_bf16<<<nFfn  / 256, 256, 0, stream>>>(pf_w2, w2_b, nFfn);
    cvt_bf16<<<(int)(MS * HID / 256), 256, 0, stream>>>(src, srcb, (int)(MS * HID));

    featurize<<<(int)M, 256, 0, stream>>>(trg, ft_w, ft_b, x, xb);

    const dim3 gProj(2, (unsigned)(M / 64));    // N=256
    const dim3 gSrc (2, (unsigned)(MS / 64));   // N=256, M=32768
    const dim3 gFfn1(8, (unsigned)(M / 64));    // N=1024

    for (int l = 0; l < N_LAYERS; ++l) {
        const size_t wo = (size_t)l * PW;
        const size_t bo = (size_t)l * HID;
        const float* g  = ln_g + bo;
        const float* be = ln_b + bo;

        // --- self attention ---
        gemm_bias<false, true ><<<gProj, 256, 0, stream>>>(xb, wq_b + wo, sa_bq + bo, qb, (int)M, HID, HID);
        gemm_bias<false, true ><<<gProj, 256, 0, stream>>>(xb, wk_b + wo, sa_bk + bo, kb, (int)M, HID, HID);
        gemm_bias<false, true ><<<gProj, 256, 0, stream>>>(xb, wv_b + wo, sa_bv + bo, vb, (int)M, HID, HID);
        attention<NA><<<dim3(NA / 16, N_HEADS, BATCH), 32, 0, stream>>>(qb, kb, vb, ab);
        gemm_bias<false, false><<<gProj, 256, 0, stream>>>(ab, wf_b + wo, sa_bf + bo, tm, (int)M, HID, HID);
        add_ln<<<(int)M, 256, 0, stream>>>(x, tm, g, be, xb);

        // --- encoder (cross) attention ---
        gemm_bias<false, true ><<<gProj, 256, 0, stream>>>(xb,   eq_b + wo, ea_bq + bo, qb, (int)M,  HID, HID);
        gemm_bias<false, true ><<<gSrc,  256, 0, stream>>>(srcb, ek_b + wo, ea_bk + bo, kb, (int)MS, HID, HID);
        gemm_bias<false, true ><<<gSrc,  256, 0, stream>>>(srcb, ev_b + wo, ea_bv + bo, vb, (int)MS, HID, HID);
        attention<NP><<<dim3(NA / 16, N_HEADS, BATCH), 32, 0, stream>>>(qb, kb, vb, ab);
        gemm_bias<false, false><<<gProj, 256, 0, stream>>>(ab, ef_b + wo, ea_bf + bo, tm, (int)M, HID, HID);
        add_ln<<<(int)M, 256, 0, stream>>>(x, tm, g, be, xb);

        // --- position-wise FFN ---
        gemm_bias<true,  true ><<<gFfn1, 256, 0, stream>>>(xb, w1_b + (size_t)l * PF_DIM * HID,
                                                           pf_b1 + (size_t)l * PF_DIM, hb, (int)M, PF_DIM, HID);
        gemm_bias<false, false><<<gProj, 256, 0, stream>>>(hb, w2_b + (size_t)l * HID * PF_DIM,
                                                           pf_b2 + bo, tm, (int)M, HID, PF_DIM);
        add_ln<<<(int)M, 256, 0, stream>>>(x, tm, g, be, xb);
    }

    pool<<<BATCH, 256, 0, stream>>>(x, pld);
    head<<<BATCH, 256, 0, stream>>>(pld, fc1_w, fc1_b, fc2_w, fc2_b, out);
}
